// DiplomacyPolicyNet_35364760715430
// MI455X (gfx1250) — matrix operand order repaired and checked by reading, not compiled, not run on hardware
//
#include <hip/hip_runtime.h>

typedef __bf16 bf16;
typedef __attribute__((ext_vector_type(16))) __bf16 v16bf;
typedef __attribute__((ext_vector_type(8)))  float  v8f;

union FragAB { v16bf v; uint4 q[2]; };
union FragC  { v8f v; float f[8]; };

__device__ inline bf16 f2bf(float f) {
  unsigned u = __builtin_bit_cast(unsigned, f);
  unsigned short h = (unsigned short)((u + 0x7FFFu + ((u >> 16) & 1u)) >> 16);
  return __builtin_bit_cast(bf16, h);
}

__device__ inline float gelu_exact(float v) {
  return 0.5f * v * (1.f + erff(v * 0.70710678118654752f));
}

// ---------------------------------------------------------------------------
// Register-blocked bf16 WMMA GEMM: each wave computes a 32x64 macro-tile
// (2 M-tiles x 4 N-tiles = 8 independent accumulators -> pipelined wmmas).
// C[M,N] = A[M,K] @ Wt[N,K]^T.  Requires M%32==0, N%64==0.
// EPI: 0 = raw f32, 1 = +bias f32, 2 = +bias,gelu f32, 3 = +bias,gelu bf16
// ---------------------------------------------------------------------------
template <int EPI, int K>
__global__ __launch_bounds__(256) void gemm_wmma_blk(
    const bf16* __restrict__ A, const bf16* __restrict__ Wt,
    const float* __restrict__ bias, float* __restrict__ Cf,
    bf16* __restrict__ Cb, int M, int N) {
  const int tilesN = N >> 6;                  // macro-tiles along N (64 wide)
  int macro = blockIdx.x * 8 + (threadIdx.x >> 5);
  int lane = threadIdx.x & 31;
  int total = (M >> 5) * tilesN;
  if (macro >= total) return;                 // wave-uniform exit
  int mt = (macro / tilesN) * 32;
  int nb = (macro % tilesN) * 64;
  int l16 = lane & 15, hi = lane >> 4;

  // Base pointers; mi/ni offsets are compile-time constants folded into
  // the 24-bit instruction offset of global_load_b128.
  const bf16* Ap = A + (size_t)(mt + l16) * K + hi * 8;    // A frag: k=g*16+hi*8+i%8
  const bf16* Wp = Wt + (size_t)(nb + l16) * K + hi * 8;   // B frag: k=g*16+hi*8+i%8

  FragC acc[2][4];
#pragma unroll
  for (int mi = 0; mi < 2; ++mi)
#pragma unroll
    for (int ni = 0; ni < 4; ++ni)
#pragma unroll
      for (int r = 0; r < 8; ++r) acc[mi][ni].f[r] = 0.f;

#pragma unroll
  for (int k0 = 0; k0 < K; k0 += 32) {
    FragAB a[2], b[4];
#pragma unroll
    for (int mi = 0; mi < 2; ++mi) {
      a[mi].q[0] = *(const uint4*)(Ap + (size_t)mi * 16 * K + k0);
      a[mi].q[1] = *(const uint4*)(Ap + (size_t)mi * 16 * K + k0 + 16);
    }
#pragma unroll
    for (int ni = 0; ni < 4; ++ni) {
      b[ni].q[0] = *(const uint4*)(Wp + (size_t)ni * 16 * K + k0);
      b[ni].q[1] = *(const uint4*)(Wp + (size_t)ni * 16 * K + k0 + 16);
    }
#pragma unroll
    for (int mi = 0; mi < 2; ++mi)
#pragma unroll
      for (int ni = 0; ni < 4; ++ni)
        acc[mi][ni].v = __builtin_amdgcn_wmma_f32_16x16x32_bf16(
            false, a[mi].v, false, b[ni].v, (short)0, acc[mi][ni].v, false, false);
  }

#pragma unroll
  for (int ni = 0; ni < 4; ++ni) {
    int n = nb + ni * 16 + l16;
    float bv = (EPI >= 1) ? bias[n] : 0.f;
#pragma unroll
    for (int mi = 0; mi < 2; ++mi) {
#pragma unroll
      for (int r = 0; r < 8; ++r) {
        int mm = mt + mi * 16 + r + 8 * hi;   // C layout: VGPR r -> row r + 8*hi
        float v = acc[mi][ni].f[r] + bv;
        if (EPI >= 2) v = gelu_exact(v);
        if (EPI == 3) Cb[(size_t)mm * N + n] = f2bf(v);
        else          Cf[(size_t)mm * N + n] = v;
      }
    }
  }
}

// Simple wave-per-16x16-tile variant (for N not divisible by 64, i.e. head2).
template <int EPI, int K>
__global__ __launch_bounds__(256) void gemm_wmma(
    const bf16* __restrict__ A, const bf16* __restrict__ Wt,
    const float* __restrict__ bias, float* __restrict__ Cf,
    bf16* __restrict__ Cb, int M, int N) {
  const int tilesN = N >> 4;
  int tile = blockIdx.x * 8 + (threadIdx.x >> 5);
  int lane = threadIdx.x & 31;
  int totalTiles = (M >> 4) * tilesN;
  if (tile >= totalTiles) return;
  int mt = tile / tilesN, nt = tile % tilesN;
  int l16 = lane & 15, hi = lane >> 4;
  const bf16* Ap = A + (size_t)(mt * 16 + l16) * K + hi * 8;
  const bf16* Wp = Wt + (size_t)(nt * 16 + l16) * K + hi * 8;
  FragC acc;
#pragma unroll
  for (int r = 0; r < 8; ++r) acc.f[r] = 0.f;
#pragma unroll
  for (int k0 = 0; k0 < K; k0 += 32) {
    FragAB a, b;
    a.q[0] = *(const uint4*)(Ap + k0);
    a.q[1] = *(const uint4*)(Ap + k0 + 16);
    b.q[0] = *(const uint4*)(Wp + k0);
    b.q[1] = *(const uint4*)(Wp + k0 + 16);
    acc.v = __builtin_amdgcn_wmma_f32_16x16x32_bf16(
        false, a.v, false, b.v, (short)0, acc.v, false, false);
  }
  int n = nt * 16 + l16;
  float bv = (EPI >= 1) ? bias[n] : 0.f;
#pragma unroll
  for (int r = 0; r < 8; ++r) {
    int mm = mt * 16 + r + 8 * hi;
    float v = acc.f[r] + bv;
    if (EPI >= 2) v = gelu_exact(v);
    if (EPI == 3) Cb[(size_t)mm * N + n] = f2bf(v);
    else          Cf[(size_t)mm * N + n] = v;
  }
}

// ---------------------------------------------------------------------------
// Elementwise / LN / attention helper kernels (f32 VALU path)
// ---------------------------------------------------------------------------
__device__ inline void ln256_store(float v, const float* g, const float* bta,
                                   float* outf, bf16* outb, size_t base, int t,
                                   float* red) {
  red[t] = v; __syncthreads();
  for (int s = 128; s > 0; s >>= 1) { if (t < s) red[t] += red[t + s]; __syncthreads(); }
  float mean = red[0] * (1.f / 256.f); __syncthreads();
  float d = v - mean;
  red[t] = d * d; __syncthreads();
  for (int s = 128; s > 0; s >>= 1) { if (t < s) red[t] += red[t + s]; __syncthreads(); }
  float var = red[0] * (1.f / 256.f); __syncthreads();
  float y = d * rsqrtf(var + 1e-5f) * g[t] + bta[t];
  if (outf) outf[base + t] = y;
  if (outb) outb[base + t] = f2bf(y);
}

__global__ void ln_rows(const float* __restrict__ resid, const float* __restrict__ delta,
                        const float* __restrict__ dbias, const float* __restrict__ g,
                        const float* __restrict__ b, float* outf, bf16* outb) {
  size_t row = blockIdx.x; int t = threadIdx.x;
  size_t idx = row * 256 + t;
  float v = delta[idx] + (resid ? resid[idx] : 0.f) + (dbias ? dbias[t] : 0.f);
  __shared__ float red[256];
  ln256_store(v, g, b, outf, outb, row * 256, t, red);
}

__global__ void wconv(const float* __restrict__ src, bf16* __restrict__ dst,
                      int K, int N, int Kp, int Np) {
  int idx = blockIdx.x * 256 + threadIdx.x;
  if (idx >= Kp * Np) return;
  int n = idx / Kp, k = idx % Kp;
  float v = (k < K && n < N) ? src[(size_t)k * N + n] : 0.f;   // transpose+pad
  dst[idx] = f2bf(v);
}

__global__ void board_pad(const float* __restrict__ board, bf16* __restrict__ out, int R) {
  int idx = blockIdx.x * 256 + threadIdx.x;
  if (idx >= R * 64) return;
  int r = idx >> 6, c = idx & 63;
  out[idx] = f2bf(c < 47 ? board[(size_t)r * 47 + c] : 0.f);
}

__global__ void s_kernel(const float* __restrict__ h, const float* __restrict__ a_src,
                         const float* __restrict__ a_dst, float* __restrict__ s_src,
                         float* __restrict__ s_dst) {
  int row = blockIdx.x; int t = threadIdx.x; int dd = t & 63;
  float hv = h[(size_t)row * 256 + t];
  __shared__ float p1[256], p2[256];
  p1[t] = hv * a_src[t]; p2[t] = hv * a_dst[t];
  __syncthreads();
  for (int s = 32; s > 0; s >>= 1) {
    if (dd < s) { p1[t] += p1[t + s]; p2[t] += p2[t + s]; }
    __syncthreads();
  }
  if (dd == 0) {
    s_src[(size_t)row * 4 + (t >> 6)] = p1[t];
    s_dst[(size_t)row * 4 + (t >> 6)] = p2[t];
  }
}

// Fused GAT: masked leaky-relu scores, softmax over j, aggregate, residual, LN1
__global__ void gat_aggr(const float* __restrict__ hbuf, const float* __restrict__ s_src,
                         const float* __restrict__ s_dst, const float* __restrict__ adj,
                         const float* __restrict__ ln_g, const float* __restrict__ ln_b,
                         float* __restrict__ x, bf16* __restrict__ xb) {
  int row = blockIdx.x;                       // b*81 + i
  int b = row / 81, i = row % 81;
  int t = threadIdx.x;
  __shared__ float e_s[81 * 4];
  __shared__ float red[256];
  __shared__ float ssrc[4];
  if (t < 4) ssrc[t] = s_src[(size_t)row * 4 + t];
  __syncthreads();
  for (int idx = t; idx < 324; idx += 256) {
    int j = idx >> 2, hh = idx & 3;
    float e = ssrc[hh] + s_dst[((size_t)b * 81 + j) * 4 + hh];
    e = e >= 0.f ? e : 0.2f * e;
    if (adj[i * 81 + j] <= 0.f) e = -1e30f;
    e_s[idx] = e;
  }
  __syncthreads();
  if (t < 4) {                                // per-head softmax over 81 neighbors
    float mx = -1e30f;
    for (int j = 0; j < 81; ++j) mx = fmaxf(mx, e_s[j * 4 + t]);
    float sm = 0.f;
    for (int j = 0; j < 81; ++j) { float a = expf(e_s[j * 4 + t] - mx); e_s[j * 4 + t] = a; sm += a; }
    float inv = 1.f / sm;
    for (int j = 0; j < 81; ++j) e_s[j * 4 + t] *= inv;
  }
  __syncthreads();
  int hh = t >> 6;
  float acc = 0.f;
  for (int j = 0; j < 81; ++j)
    acc += e_s[j * 4 + hh] * hbuf[((size_t)b * 81 + j) * 256 + t];
  float v = x[(size_t)row * 256 + t] + acc;
  ln256_store(v, ln_g, ln_b, x, xb, (size_t)row * 256, t, red);
}

__global__ void ctx_kernel(float* __restrict__ x, bf16* __restrict__ xb,
                           const float* __restrict__ pemb, const int* __restrict__ pidx) {
  int row = blockIdx.x; int b = row / 81; int t = threadIdx.x;
  size_t idx = (size_t)row * 256 + t;
  float v = x[idx] + pemb[(size_t)pidx[b] * 256 + t];
  x[idx] = v; xb[idx] = f2bf(v);
}

__global__ void gather_ue(const float* __restrict__ ctx, const int* __restrict__ uidx,
                          float* __restrict__ ue, bf16* __restrict__ ueb) {
  int row = blockIdx.x; int t = threadIdx.x;
  int b = row / 17, u = row % 17;
  int n = uidx[b * 17 + u]; if (n < 0) n = 0;
  float v = ctx[((size_t)b * 81 + n) * 256 + t];
  ue[(size_t)row * 256 + t] = v;
  ueb[(size_t)row * 256 + t] = f2bf(v);
}

// Fused decode attention: scores, softmax over 81, aggregate V, residual, LN
__global__ void attn_kernel(const float* __restrict__ Qf, const float* __restrict__ Kc,
                            const float* __restrict__ Vc, const float* __restrict__ ue,
                            const float* __restrict__ g, const float* __restrict__ bta,
                            bf16* __restrict__ urb) {
  int row = blockIdx.x; int b = row / 17;
  int t = threadIdx.x;
  __shared__ float q[256]; __shared__ float sc[81]; __shared__ float red[256];
  q[t] = Qf[(size_t)row * 256 + t];
  __syncthreads();
  if (t < 81) {
    const float* kr = Kc + ((size_t)b * 81 + t) * 256;
    float a = 0.f;
    for (int d2 = 0; d2 < 256; ++d2) a += q[d2] * kr[d2];
    sc[t] = a * 0.0625f;                      // 1/sqrt(256)
  }
  __syncthreads();
  if (t == 0) {
    float mx = -1e30f;
    for (int j = 0; j < 81; ++j) mx = fmaxf(mx, sc[j]);
    float sm = 0.f;
    for (int j = 0; j < 81; ++j) { float e = expf(sc[j] - mx); sc[j] = e; sm += e; }
    float inv = 1.f / sm;
    for (int j = 0; j < 81; ++j) sc[j] *= inv;
  }
  __syncthreads();
  float acc = 0.f;
  for (int j = 0; j < 81; ++j) acc += sc[j] * Vc[((size_t)b * 81 + j) * 256 + t];
  float v = ue[(size_t)row * 256 + t] + acc;
  ln256_store(v, g, bta, nullptr, urb, (size_t)row * 256, t, red);
}

__global__ void final_kernel(const float* __restrict__ c, const float* __restrict__ b2,
                             float* __restrict__ out, int total) {
  int idx = blockIdx.x * 256 + threadIdx.x;
  if (idx >= total) return;
  int v = idx % 169; size_t r = idx / 169;
  out[idx] = c[r * 176 + v] + b2[v];
}

// ---------------------------------------------------------------------------
extern "C" void kernel_launch(void* const* d_in, const int* in_sizes, int n_in,
                              void* d_out, int out_size, void* d_ws, size_t ws_size,
                              hipStream_t stream) {
  (void)in_sizes; (void)n_in; (void)out_size; (void)ws_size;
  const int B = 1024, Nn = 81, U = 17;
  const int BN = B * Nn;        // 82944 (multiple of 32)
  const int BU = B * U;         // 17408 (multiple of 32)

  auto F = [&](int i) { return (const float*)d_in[i]; };
  // pytree (sorted-dict) flatten order of setup_inputs():
  const float* board = F(0);
  const float* adj = F(1);
  const int* unit_indices = (const int*)d_in[2];
  const int* power_indices = (const int*)d_in[3];
  const float* attn_ln_b = F(4); const float* attn_ln_g = F(5);
  struct Blk { const float *W, *a_dst, *a_src, *ffn_b1, *ffn_b2, *ffn_w1, *ffn_w2,
                           *ln1_b, *ln1_g, *ln2_b, *ln2_g; } blk[3];
  for (int l = 0; l < 3; ++l) {
    int o = 6 + 11 * l;
    blk[l] = { F(o), F(o+1), F(o+2), F(o+3), F(o+4), F(o+5), F(o+6), F(o+7), F(o+8), F(o+9), F(o+10) };
  }
  const float* head_b1 = F(39); const float* head_b2 = F(40);
  const float* head_ln_b = F(41); const float* head_ln_g = F(42);
  const float* head_w1 = F(43); const float* head_w2 = F(44);
  const float* in_b = F(45); const float* in_ln_b = F(46); const float* in_ln_g = F(47);
  const float* in_w = F(48);
  const float* k_b = F(49); const float* k_w = F(50);
  const float* power_embed = F(51);
  const float* q_b = F(52); const float* q_w = F(53);
  const float* v_b = F(54); const float* v_w = F(55);

  // ---- workspace carving ----
  char* ws = (char*)d_ws;
  size_t off = 0;
  auto carve = [&](size_t bytes) -> char* {
    char* p = ws + off; off += (bytes + 255) & ~(size_t)255; return p;
  };
  float* x   = (float*)carve((size_t)BN * 256 * 4);   // activations f32
  bf16*  xb  = (bf16*)carve((size_t)BN * 256 * 2);    // bf16 mirror
  float* hG  = (float*)carve((size_t)BN * 256 * 4);   // GAT features / Kc
  float* t0  = (float*)carve((size_t)BN * 256 * 4);   // GEMM scratch / Vc
  char*  arena = carve((size_t)BN * 512 * 2);         // FFN hidden / boardb / decode
  float* ssrc = (float*)carve((size_t)BN * 4 * 4);
  float* sdst = (float*)carve((size_t)BN * 4 * 4);
  bf16*  wp   = (bf16*)carve((size_t)2800000);        // bf16 weight arena

  // arena aliases (phases never overlap in time)
  bf16* boardb = (bf16*)arena;                        // input stage only
  bf16* c1bf   = (bf16*)arena;                        // FFN hidden, per layer
  char* ar = arena;                                   // decode stage carve
  float* ue  = (float*)ar; ar += (size_t)BU * 256 * 4;
  bf16*  ueb = (bf16*)ar;  ar += (size_t)BU * 256 * 2;
  bf16*  urb = (bf16*)ar;  ar += (size_t)BU * 256 * 2;
  float* t1  = (float*)ar; ar += (size_t)BU * 256 * 4;
  bf16*  hbf = (bf16*)ar;  ar += (size_t)BU * 256 * 2;
  float* Qf  = (float*)ar; ar += (size_t)BU * 256 * 4;
  float* c2  = Qf;                                    // head2 out reuses Q region

  // bf16 weight pointers (transposed to [N,K], padded)
  bf16* w_in = wp; wp += 256 * 64;
  bf16 *wWt[3], *w1t[3], *w2t[3];
  for (int l = 0; l < 3; ++l) {
    wWt[l] = wp; wp += 256 * 256;
    w1t[l] = wp; wp += 512 * 256;
    w2t[l] = wp; wp += 256 * 512;
  }
  bf16* qwt  = wp; wp += 256 * 256;
  bf16* kwt  = wp; wp += 256 * 256;
  bf16* vwt  = wp; wp += 256 * 256;
  bf16* hw1t = wp; wp += 256 * 256;
  bf16* hw2t = wp; wp += 176 * 256;

  auto wcv = [&](const float* src, bf16* dst, int K, int N2, int Kp, int Np) {
    int total = Kp * Np;
    wconv<<<(total + 255) / 256, 256, 0, stream>>>(src, dst, K, N2, Kp, Np);
  };
  auto gBb = [](int M, int N2) { return ((M >> 5) * (N2 >> 6) + 7) / 8; };  // blocked
  auto gB  = [](int M, int N2) { return ((M >> 4) * (N2 >> 4) + 7) / 8; };  // simple

  // ---- weight conversion (f32 -> transposed bf16) ----
  wcv(in_w, w_in, 47, 256, 64, 256);
  for (int l = 0; l < 3; ++l) {
    wcv(blk[l].W, wWt[l], 256, 256, 256, 256);
    wcv(blk[l].ffn_w1, w1t[l], 256, 512, 256, 512);
    wcv(blk[l].ffn_w2, w2t[l], 512, 256, 512, 256);
  }
  wcv(q_w, qwt, 256, 256, 256, 256);
  wcv(k_w, kwt, 256, 256, 256, 256);
  wcv(v_w, vwt, 256, 256, 256, 256);
  wcv(head_w1, hw1t, 256, 256, 256, 256);
  wcv(head_w2, hw2t, 256, 169, 256, 176);

  // ---- input projection: GELU(board@in_w + in_b) -> LN ----
  board_pad<<<((BN * 64) + 255) / 256, 256, 0, stream>>>(board, boardb, BN);
  gemm_wmma_blk<2, 64><<<gBb(BN, 256), 256, 0, stream>>>(boardb, w_in, in_b, t0, nullptr, BN, 256);
  ln_rows<<<BN, 256, 0, stream>>>(nullptr, t0, nullptr, in_ln_g, in_ln_b, x, xb);

  // ---- 3 GAT blocks ----
  for (int l = 0; l < 3; ++l) {
    gemm_wmma_blk<0, 256><<<gBb(BN, 256), 256, 0, stream>>>(xb, wWt[l], nullptr, hG, nullptr, BN, 256);
    s_kernel<<<BN, 256, 0, stream>>>(hG, blk[l].a_src, blk[l].a_dst, ssrc, sdst);
    gat_aggr<<<BN, 256, 0, stream>>>(hG, ssrc, sdst, adj, blk[l].ln1_g, blk[l].ln1_b, x, xb);
    gemm_wmma_blk<3, 256><<<gBb(BN, 512), 256, 0, stream>>>(xb, w1t[l], blk[l].ffn_b1, nullptr, c1bf, BN, 512);
    gemm_wmma_blk<0, 512><<<gBb(BN, 256), 256, 0, stream>>>(c1bf, w2t[l], nullptr, t0, nullptr, BN, 256);
    ln_rows<<<BN, 256, 0, stream>>>(x, t0, blk[l].ffn_b2, blk[l].ln2_g, blk[l].ln2_b, x, xb);
  }

  // ---- decode ----
  ctx_kernel<<<BN, 256, 0, stream>>>(x, xb, power_embed, power_indices);
  gather_ue<<<BU, 256, 0, stream>>>(x, unit_indices, ue, ueb);
  gemm_wmma_blk<1, 256><<<gBb(BU, 256), 256, 0, stream>>>(ueb, qwt, q_b, Qf, nullptr, BU, 256);
  gemm_wmma_blk<1, 256><<<gBb(BN, 256), 256, 0, stream>>>(xb, kwt, k_b, hG, nullptr, BN, 256);
  gemm_wmma_blk<1, 256><<<gBb(BN, 256), 256, 0, stream>>>(xb, vwt, v_b, t0, nullptr, BN, 256);
  attn_kernel<<<BU, 256, 0, stream>>>(Qf, hG, t0, ue, attn_ln_g, attn_ln_b, urb);
  gemm_wmma_blk<2, 256><<<gBb(BU, 256), 256, 0, stream>>>(urb, hw1t, head_b1, t1, nullptr, BU, 256);
  ln_rows<<<BU, 256, 0, stream>>>(nullptr, t1, nullptr, head_ln_g, head_ln_b, nullptr, hbf);
  gemm_wmma<0, 256><<<gB(BU, 176), 256, 0, stream>>>(hbf, hw2t, nullptr, c2, nullptr, BU, 176);
  final_kernel<<<((BU * 169) + 255) / 256, 256, 0, stream>>>(c2, head_b2, (float*)d_out, BU * 169);
}
